// LSTMFromScratch_33930241638835
// MI455X (gfx1250) — matrix operand order repaired and checked by reading, not compiled, not run on hardware
//
#include <hip/hip_runtime.h>
#include <hip/hip_bf16.h>
#include <stdint.h>

typedef __attribute__((ext_vector_type(16))) __bf16 v16bf;
typedef __attribute__((ext_vector_type(8)))  float  v8f;

#define BATCH  128
#define SEQ    256
#define INPUT  512
#define HIDDEN 1024
#define OUTPUT 512
#define NBLK_SCAN 64          // one block per 16-wide hidden tile
#define LDP 1032              // padded row stride (elements) -> 516 dwords, 516%64=4

// ---------- helpers ----------
__device__ __forceinline__ unsigned short f2bf(float f) {
  union { float f; uint32_t u; } v; v.f = f;
  uint32_t u = v.u;
  uint32_t r = (u + 0x7FFFu + ((u >> 16) & 1u)) >> 16;   // round-to-nearest-even
  return (unsigned short)r;
}
__device__ __forceinline__ float sigmoid_(float x) { return 1.0f / (1.0f + __expf(-x)); }
__device__ __forceinline__ float tanh_(float x)    { return 1.0f - 2.0f / (__expf(2.0f * x) + 1.0f); }

union FragU { v16bf v; uint4 q[2]; };

// A-matrix 16x32 bf16 fragment: lanes 0-15 hold K[0..15];
// lanes 16-31 hold K[16..31].  Row-major A, stride lda (elements).
__device__ __forceinline__ v16bf load_a_frag(const unsigned short* A, int lda,
                                             int mBase, int kBase, int lane) {
  int hf = (lane >> 4) & 1, row = lane & 15;
  const unsigned short* p = A + (size_t)(mBase + row) * lda + kBase + 16 * hf;
  FragU f;
  f.q[0] = *(const uint4*)(p);        // K = kBase + 16*hf + [0..7]
  f.q[1] = *(const uint4*)(p + 8);    // K = kBase + 16*hf + [8..15]
  return f.v;
}

// B-matrix 32x16 bf16 fragment: lanes 0-15 hold K[0..15], lanes 16-31 K[16..31],
// column n = lane&15.  W row-major [n][k] (out = A * W^T).
__device__ __forceinline__ v16bf load_b_frag(const unsigned short* W, int ldw,
                                             int nBase, int kBase, int lane) {
  int hf = (lane >> 4) & 1, col = lane & 15;
  const unsigned short* p = W + (size_t)(nBase + col) * ldw + kBase + 16 * hf;
  FragU f;
  f.q[0] = *(const uint4*)(p);
  f.q[1] = *(const uint4*)(p + 8);
  return f.v;
}

// Same B fragment out of the LDS weight panel (row stride LDP, padded vs bank conflicts).
__device__ __forceinline__ v16bf load_b_lds(const unsigned short* panel_g, int kBase, int lane) {
  int hf = (lane >> 4) & 1, col = lane & 15;
  const unsigned short* p = panel_g + (size_t)col * LDP + kBase + 16 * hf;
  FragU f;
  f.q[0] = *(const uint4*)(p);
  f.q[1] = *(const uint4*)(p + 8);
  return f.v;
}

// ---------- conversion / setup kernels ----------
__global__ void k_f2bf(const float* __restrict__ src, unsigned short* __restrict__ dst, int n) {
  for (int i = blockIdx.x * blockDim.x + threadIdx.x; i < n; i += gridDim.x * blockDim.x)
    dst[i] = f2bf(src[i]);
}

__global__ void k_biascat(const float* __restrict__ bf, const float* __restrict__ bi,
                          const float* __restrict__ bg, const float* __restrict__ bo,
                          float* __restrict__ dst) {
  int t = blockIdx.x * blockDim.x + threadIdx.x;
  if (t < HIDDEN) {
    dst[t] = bf[t]; dst[HIDDEN + t] = bi[t];
    dst[2 * HIDDEN + t] = bg[t]; dst[3 * HIDDEN + t] = bo[t];
  }
}

// ---------- input projection: x (32768x512) @ Wx^T (512x4096) + bias ----------
__global__ void k_xproj(const unsigned short* __restrict__ xbf,
                        const unsigned short* __restrict__ Wx,     // [4096][512] bf16
                        const float* __restrict__ bias_cat,        // [4096]
                        float* __restrict__ xproj) {
  int wid  = (blockIdx.x * blockDim.x + threadIdx.x) >> 5;  // 0..131071
  int lane = threadIdx.x & 31;
  int nblk  = wid >> 11;       // 0..63  -> waves in a block share the B panel (WGP$ hits)
  int mtile = wid & 2047;      // 0..2047
  int mBase = mtile * 16, nBase = nblk * 64;

  v8f acc[4] = {};
  for (int k = 0; k < INPUT; k += 32) {
    v16bf a = load_a_frag(xbf, INPUT, mBase, k, lane);
#pragma unroll
    for (int j = 0; j < 4; ++j) {
      v16bf b = load_b_frag(Wx, INPUT, nBase + 16 * j, k, lane);
      acc[j] = __builtin_amdgcn_wmma_f32_16x16x32_bf16(false, a, false, b,
                                                       (short)0, acc[j], false, false);
    }
  }
  int hf = (lane >> 4) & 1, col = lane & 15;
#pragma unroll
  for (int j = 0; j < 4; ++j) {
    int n = nBase + 16 * j + col;
    int gate = n >> 10, hh = n & 1023;
    float bv = bias_cat[n];
#pragma unroll
    for (int r = 0; r < 8; ++r) {
      int m = mBase + r + 8 * hf;          // row = b*SEQ + s
      int bb = m >> 8, ss = m & 255;
      xproj[(((size_t)ss * 4 + gate) * BATCH + bb) * HIDDEN + hh] = acc[j][r] + bv;
    }
  }
}

// ---------- device-wide barrier (monotonic counter, release/acquire fences) ----------
__device__ __forceinline__ void grid_barrier(unsigned* cnt, int step) {
  __builtin_amdgcn_fence(__ATOMIC_RELEASE, "agent");   // flush this wave's stores
  __syncthreads();                                     // whole block arrived (all waves fenced)
  if (threadIdx.x == 0) {
    __hip_atomic_fetch_add(cnt, 1u, __ATOMIC_RELAXED, __HIP_MEMORY_SCOPE_AGENT);
    unsigned target = (unsigned)(step + 1) * NBLK_SCAN;
    while (__hip_atomic_load(cnt, __ATOMIC_RELAXED, __HIP_MEMORY_SCOPE_AGENT) < target) {
      __builtin_amdgcn_s_sleep(2);
    }
  }
  __syncthreads();
  __builtin_amdgcn_fence(__ATOMIC_ACQUIRE, "agent");   // invalidate stale cache lines
}

// ---------- persistent scan kernel: all 256 timesteps in one launch ----------
// Block b owns hidden tile [16b,16b+16) for all 4 gates; its 8 waves cover the 8
// batch tiles.  The 128KB weight panel (4 gates x 16 cols x 1024 K, bf16) lives in
// LDS for the whole scan (CDNA5: 320KB LDS/WGP).  h_bf16 is double-buffered across
// the grid barrier; c is owned per (b,h) so it updates in place.
__global__ void k_scan(const unsigned short* __restrict__ Wh,      // [4][1024][1024] bf16
                       const float* __restrict__ xproj,            // [S][4][128][1024]
                       float* __restrict__ c, float* __restrict__ h,
                       unsigned short* __restrict__ hbfA,
                       unsigned short* __restrict__ hbfB,
                       unsigned* __restrict__ barcnt) {
  __shared__ unsigned short panel[4 * 16 * LDP];       // 132096 bytes
  int htile = blockIdx.x;            // 0..63
  int wave  = threadIdx.x >> 5;      // 0..7  = batch tile
  int lane  = threadIdx.x & 31;
  int hBase = htile * 16, mBase = wave * 16;

  // ---- stage the weight panel into LDS once (16B chunks, all 256 threads) ----
  for (int idx = threadIdx.x; idx < 4 * 16 * (HIDDEN / 8); idx += 256) {
    int g   = idx >> 11;             // 2048 chunks per gate
    int rem = idx & 2047;
    int col = rem >> 7;              // 128 chunks per row
    int kc  = (rem & 127) * 8;
    const unsigned short* src = Wh + ((size_t)g * HIDDEN + hBase + col) * HIDDEN + kc;
    *(uint4*)&panel[((size_t)g * 16 + col) * LDP + kc] = *(const uint4*)src;
  }
  __syncthreads();

  int hf = (lane >> 4) & 1, col = lane & 15;
  int hh = hBase + col;
  const size_t GS = (size_t)BATCH * HIDDEN;

#pragma unroll 1
  for (int t = 0; t < SEQ; ++t) {
    const unsigned short* hin  = (t & 1) ? hbfB : hbfA;
    unsigned short*       hout = (t & 1) ? hbfA : hbfB;
    const float* xp = xproj + (size_t)t * 4 * GS;

    v8f acc[4] = {};                 // f, i, g, o for the same 16x16 tile
    for (int k = 0; k < HIDDEN; k += 32) {
      v16bf a = load_a_frag(hin, HIDDEN, mBase, k, lane);
#pragma unroll
      for (int g = 0; g < 4; ++g) {
        v16bf b = load_b_lds(&panel[(size_t)g * 16 * LDP], k, lane);
        acc[g] = __builtin_amdgcn_wmma_f32_16x16x32_bf16(false, a, false, b,
                                                         (short)0, acc[g], false, false);
      }
    }
#pragma unroll
    for (int r = 0; r < 8; ++r) {
      int bb = mBase + r + 8 * hf;
      size_t base = (size_t)bb * HIDDEN + hh;
      float fv = sigmoid_(acc[0][r] + xp[0 * GS + base]);
      float iv = sigmoid_(acc[1][r] + xp[1 * GS + base]);
      float gv = tanh_   (acc[2][r] + xp[2 * GS + base]);
      float ov = sigmoid_(acc[3][r] + xp[3 * GS + base]);
      float cn = fv * c[base] + iv * gv;
      float hn = ov * tanh_(cn);
      c[base] = cn;
      if (t == SEQ - 1) h[base] = hn;
      hout[base] = f2bf(hn);
    }
    grid_barrier(barcnt, t);         // h handoff between timesteps
  }
}

// ---------- final projection: h (128x1024) @ Why^T (1024x512) + bias ----------
__global__ void k_out(const unsigned short* __restrict__ hbf,
                      const unsigned short* __restrict__ Wy,   // [512][1024] bf16
                      const float* __restrict__ yb, float* __restrict__ out) {
  int wid  = (blockIdx.x * blockDim.x + threadIdx.x) >> 5;  // 0..255
  int lane = threadIdx.x & 31;
  int ntile = wid >> 3;        // 0..31 (waves in a block share B)
  int mtile = wid & 7;         // 0..7
  int mBase = mtile * 16, nBase = ntile * 16;

  v8f acc = {};
  for (int k = 0; k < HIDDEN; k += 32) {
    v16bf a = load_a_frag(hbf, HIDDEN, mBase, k, lane);
    v16bf b = load_b_frag(Wy, HIDDEN, nBase, k, lane);
    acc = __builtin_amdgcn_wmma_f32_16x16x32_bf16(false, a, false, b,
                                                  (short)0, acc, false, false);
  }
  int hf = (lane >> 4) & 1, col = lane & 15;
  int n = nBase + col;
  float bv = yb[n];
#pragma unroll
  for (int r = 0; r < 8; ++r) {
    int m = mBase + r + 8 * hf;
    out[(size_t)m * OUTPUT + n] = acc[r] + bv;
  }
}

__global__ void k_fincopy(const float* __restrict__ h, const float* __restrict__ c,
                          float* __restrict__ out_h, float* __restrict__ out_c) {
  int t = blockIdx.x * blockDim.x + threadIdx.x;
  if (t < BATCH * HIDDEN) { out_h[t] = h[t]; out_c[t] = c[t]; }
}

// ---------- host ----------
extern "C" void kernel_launch(void* const* d_in, const int* in_sizes, int n_in,
                              void* d_out, int out_size, void* d_ws, size_t ws_size,
                              hipStream_t stream) {
  const float* x_seq = (const float*)d_in[0];
  const float* Wxf_w = (const float*)d_in[1];
  const float* Wxf_b = (const float*)d_in[2];
  const float* Whf_w = (const float*)d_in[3];
  const float* Wxi_w = (const float*)d_in[4];
  const float* Wxi_b = (const float*)d_in[5];
  const float* Whi_w = (const float*)d_in[6];
  const float* Wxg_w = (const float*)d_in[7];
  const float* Wxg_b = (const float*)d_in[8];
  const float* Whg_w = (const float*)d_in[9];
  const float* Wxo_w = (const float*)d_in[10];
  const float* Wxo_b = (const float*)d_in[11];
  const float* Who_w = (const float*)d_in[12];
  const float* Why_w = (const float*)d_in[13];
  const float* Why_b = (const float*)d_in[14];

  char* ws = (char*)d_ws;
  size_t off = 0;
  auto take = [&](size_t bytes) -> void* {
    void* p = ws + off;
    off += (bytes + 255) & ~(size_t)255;
    return p;
  };

  unsigned short* xbf   = (unsigned short*)take((size_t)BATCH * SEQ * INPUT * 2);
  unsigned short* Wxbf  = (unsigned short*)take((size_t)4 * HIDDEN * INPUT * 2);
  unsigned short* Whbf  = (unsigned short*)take((size_t)4 * HIDDEN * HIDDEN * 2);
  unsigned short* Wybf  = (unsigned short*)take((size_t)OUTPUT * HIDDEN * 2);
  float*          biasc = (float*)take((size_t)4 * HIDDEN * 4);
  float*          xproj = (float*)take((size_t)SEQ * 4 * BATCH * HIDDEN * 4);
  float*          cbuf  = (float*)take((size_t)BATCH * HIDDEN * 4);
  float*          hbuf  = (float*)take((size_t)BATCH * HIDDEN * 4);
  unsigned short* hbfA  = (unsigned short*)take((size_t)BATCH * HIDDEN * 2);
  unsigned short* hbfB  = (unsigned short*)take((size_t)BATCH * HIDDEN * 2);
  unsigned*       barcnt = (unsigned*)take(256);
  (void)ws_size; (void)n_in; (void)in_sizes; (void)out_size;

  // --- bf16 conversions ---
  k_f2bf<<<2048, 256, 0, stream>>>(x_seq, xbf, BATCH * SEQ * INPUT);
  const int WXN = HIDDEN * INPUT, WHN = HIDDEN * HIDDEN;
  k_f2bf<<<512, 256, 0, stream>>>(Wxf_w, Wxbf + 0 * (size_t)WXN, WXN);
  k_f2bf<<<512, 256, 0, stream>>>(Wxi_w, Wxbf + 1 * (size_t)WXN, WXN);
  k_f2bf<<<512, 256, 0, stream>>>(Wxg_w, Wxbf + 2 * (size_t)WXN, WXN);
  k_f2bf<<<512, 256, 0, stream>>>(Wxo_w, Wxbf + 3 * (size_t)WXN, WXN);
  k_f2bf<<<1024, 256, 0, stream>>>(Whf_w, Whbf + 0 * (size_t)WHN, WHN);
  k_f2bf<<<1024, 256, 0, stream>>>(Whi_w, Whbf + 1 * (size_t)WHN, WHN);
  k_f2bf<<<1024, 256, 0, stream>>>(Whg_w, Whbf + 2 * (size_t)WHN, WHN);
  k_f2bf<<<1024, 256, 0, stream>>>(Who_w, Whbf + 3 * (size_t)WHN, WHN);
  k_f2bf<<<512, 256, 0, stream>>>(Why_w, Wybf, OUTPUT * HIDDEN);
  k_biascat<<<(HIDDEN + 255) / 256, 256, 0, stream>>>(Wxf_b, Wxi_b, Wxg_b, Wxo_b, biasc);

  // --- initial state: h0 = c0 = 0; barrier counter = 0 ---
  hipMemsetAsync(cbuf, 0, (size_t)BATCH * HIDDEN * 4, stream);
  hipMemsetAsync(hbfA, 0, (size_t)BATCH * HIDDEN * 2, stream);
  hipMemsetAsync(barcnt, 0, 256, stream);

  // --- big input-projection GEMM (131072 waves, WMMA bf16) ---
  k_xproj<<<16384, 256, 0, stream>>>(xbf, Wxbf, biasc, xproj);

  // --- persistent scan: one launch for all 256 timesteps ---
  k_scan<<<NBLK_SCAN, 256, 0, stream>>>(Whbf, xproj, cbuf, hbuf, hbfA, hbfB, barcnt);

  // --- outputs: out = h @ Why^T + b ; then copy final h, c ---
  // SEQ even -> last step (t=255, odd) wrote hbfA
  float* out   = (float*)d_out;
  float* out_h = out + (size_t)BATCH * OUTPUT;
  float* out_c = out_h + (size_t)BATCH * HIDDEN;
  k_out<<<32, 256, 0, stream>>>(hbfA, Wybf, Why_b, out);
  k_fincopy<<<512, 256, 0, stream>>>(hbuf, cbuf, out_h, out_c);
}
